// Detection_65326452572229
// MI455X (gfx1250) — compile-verified, matrix operand order checked by reading, not run on hardware
//
#include <hip/hip_runtime.h>
#include <math.h>

// ---------------- CDNA5 vector types ----------------
typedef __attribute__((ext_vector_type(16))) _Float16     v16h;
typedef __attribute__((ext_vector_type(8)))  float        v8f;
typedef __attribute__((ext_vector_type(4)))  unsigned int v4u;
typedef __attribute__((ext_vector_type(8)))  unsigned int v8u;

// ---------------- problem constants ----------------
#define Bq     8
#define Nq     30000
#define Cq     81
#define NCLS   80            // classes 1..80 (skip background)
#define KPRE   300
#define TOPK   200
#define MINSC  0.05f
#define MAXIOU 0.45f
#define TOT    (Bq * Nq)     // 240000 rows

#define CH     4096          // TDM chunk (floats) per stage buffer
#define CAP    1024          // per-class candidate cap (bitonic size)
#define CAP2   512           // per-batch final candidate cap

// =====================================================================
// Kernel 1: fused softmax (WMMA row-sum on the matrix pipe) + transposed
//           probability write + box decode.
// 256 threads = 8 waves; each wave owns 16 rows. Lane L: row m=L%16,
// K-half h=L/16 (matches 16-bit A-matrix 16x32 layout: row m lives in
// lanes m and m+16).
// =====================================================================
__global__ __launch_bounds__(256) void softmax_decode_kernel(
    const float* __restrict__ scores,   // [B,N,81]
    const float* __restrict__ locs,     // [B,N,4]
    const float* __restrict__ priors,   // [N,4]
    float* __restrict__ probsT,         // [B,80,N]
    float* __restrict__ decoded)        // [B,N,4] xyxy
{
  const int wave = threadIdx.x >> 5;
  const int lane = threadIdx.x & 31;
  const int m = lane & 15;
  const int h = lane >> 4;

  const int row  = (blockIdx.x * 8 + wave) * 16 + m;
  const int rowc = (row < TOT) ? row : (TOT - 1);       // clamp for loads
  const float* rp = scores + (size_t)rowc * Cq;

  // Load 81 logits per row split across lane pair (m, m+16); pad to 96.
  float ev[3][16];
  float lmax = -1e30f;
#pragma unroll
  for (int j = 0; j < 3; ++j) {
    const int cbase = j * 32 + h * 16;
#pragma unroll
    for (int i = 0; i < 16; ++i) {
      const int c = cbase + i;
      const float v = (c < Cq) ? rp[c] : -1e30f;
      ev[j][i] = v;
      lmax = fmaxf(lmax, v);
    }
  }
  const float rmax = fmaxf(lmax, __shfl_xor(lmax, 16));

  // exp(x - max); keep f32 for the store, pack f16 for the matrix pipe.
  v16h a0, a1, a2;
#pragma unroll
  for (int i = 0; i < 16; ++i) {
    const int c0 = 0  + h * 16 + i;
    const int c1 = 32 + h * 16 + i;
    const int c2 = 64 + h * 16 + i;
    float e0 = (c0 < Cq) ? __expf(ev[0][i] - rmax) : 0.0f;
    float e1 = (c1 < Cq) ? __expf(ev[1][i] - rmax) : 0.0f;
    float e2 = (c2 < Cq) ? __expf(ev[2][i] - rmax) : 0.0f;
    ev[0][i] = e0; ev[1][i] = e1; ev[2][i] = e2;
    a0[i] = (_Float16)e0; a1[i] = (_Float16)e1; a2[i] = (_Float16)e2;
  }

  v16h ones;
#pragma unroll
  for (int i = 0; i < 16; ++i) ones[i] = (_Float16)1.0f;

  // D = A * ones + C  => every column of D holds the row sum.
  v8f acc = {};
  acc = __builtin_amdgcn_wmma_f32_16x16x32_f16(false, a0, false, ones, (short)0, acc, false, false);
  acc = __builtin_amdgcn_wmma_f32_16x16x32_f16(false, a1, false, ones, (short)0, acc, false, false);
  acc = __builtin_amdgcn_wmma_f32_16x16x32_f16(false, a2, false, ones, (short)0, acc, false, false);

  // C/D layout: VGPR r -> row r (lanes 0-15) / row r+8 (lanes 16-31).
  const int r = m & 7;
  const float s4 = (r & 4) ? ((r & 2) ? ((r & 1) ? acc[7] : acc[6])
                                      : ((r & 1) ? acc[5] : acc[4]))
                           : ((r & 2) ? ((r & 1) ? acc[3] : acc[2])
                                      : ((r & 1) ? acc[1] : acc[0]));
  const float so = __shfl_xor(s4, 16);
  const float rowsum = ((m >> 3) == h) ? s4 : so;
  const float inv = 1.0f / rowsum;

  if (row < TOT) {
    const int b = row / Nq;
    const int n = row % Nq;
    float* pb = probsT + (size_t)b * NCLS * Nq + n;
#pragma unroll
    for (int j = 0; j < 3; ++j) {
#pragma unroll
      for (int i = 0; i < 16; ++i) {
        const int c = j * 32 + h * 16 + i;
        if (c >= 1 && c < Cq) pb[(size_t)(c - 1) * Nq] = ev[j][i] * inv;
      }
    }
    if (h == 0) {  // one lane per row decodes the box
      const float* l = locs   + (size_t)row * 4;
      const float* p = priors + (size_t)n   * 4;
      const float cx = p[0] + l[0] * 0.1f * p[2];
      const float cy = p[1] + l[1] * 0.1f * p[3];
      const float w  = p[2] * __expf(l[2] * 0.2f);
      const float hh = p[3] * __expf(l[3] * 0.2f);
      float* d = decoded + (size_t)row * 4;
      d[0] = cx - 0.5f * w;  d[1] = cy - 0.5f * hh;
      d[2] = cx + 0.5f * w;  d[3] = cy + 0.5f * hh;
    }
  }
}

// =====================================================================
// Kernel 2: per-(b,c) candidate selection + bitonic sort + greedy NMS.
// The 30000-float probability row is streamed through two LDS stages by
// the Tensor Data Mover (double-buffered, s_wait_tensorcnt pipelined).
// =====================================================================
__device__ __forceinline__ void tdm_issue(const float* src, int chunk,
                                          unsigned ldsBase, int par)
{
  const unsigned long long ga = (unsigned long long)(const void*)(src + chunk * CH);
  const unsigned td0 = (unsigned)(Nq - chunk * CH);   // OOB reads return 0
  // D# group 0: count=1, lds_addr, 57-bit global_addr, type=2 ("image")
  v4u g0 = { 1u,
             ldsBase + (unsigned)(par * CH * 4),
             (unsigned)ga,
             (unsigned)((ga >> 32) & 0x1FFFFFFull) | (2u << 30) };
  // D# group 1: data_size=4B, tensor 1 x td0, tile 1 x CH, stride CH
  v8u g1 = { (2u << 16),                          // data_size = 2 (4 bytes)
             (td0 & 0xFFFFu) << 16,               // tensor_dim0 lo
             ((td0 >> 16) & 0xFFFFu) | (1u << 16),// tensor_dim0 hi | tensor_dim1=1
             ((unsigned)CH << 16),                // tile_dim0
             1u,                                  // tile_dim1 = 1
             (unsigned)CH,                        // tensor_dim0_stride
             0u, 0u };
  asm volatile("tensor_load_to_lds %0, %1" :: "s"(g0), "s"(g1) : "memory");
}

__global__ __launch_bounds__(256) void select_nms_kernel(
    const float* __restrict__ probsT,    // [B,80,N]
    const float* __restrict__ decoded,   // [B,N,4]
    float* __restrict__ keptScore,       // [B*80, KPRE]
    int*   __restrict__ keptIdx)         // [B*80, KPRE]
{
  __shared__ float stage[2][CH];
  __shared__ float cs[CAP];
  __shared__ int   ci[CAP];
  __shared__ float bx0[KPRE], by0[KPRE], bx1[KPRE], by1[KPRE], barea[KPRE];
  __shared__ int   sup[KPRE];
  __shared__ int   cnt;

  const int bc = blockIdx.x;            // 0..639
  const int b  = bc / NCLS;
  const float* src = probsT + (size_t)bc * Nq;

  if (threadIdx.x == 0) cnt = 0;
  const unsigned ldsBase = (unsigned)(size_t)(&stage[0][0]);

  const int NCHK = (Nq + CH - 1) / CH;  // 8
  if (threadIdx.x < 32) tdm_issue(src, 0, ldsBase, 0);

  for (int ch = 0; ch < NCHK; ++ch) {
    const int par = ch & 1;
    if (threadIdx.x < 32) {
      if (ch + 1 < NCHK) {
        tdm_issue(src, ch + 1, ldsBase, par ^ 1);
        __builtin_amdgcn_s_wait_tensorcnt(1);   // chunk ch landed
      } else {
        __builtin_amdgcn_s_wait_tensorcnt(0);
      }
    }
    __syncthreads();                            // data visible to all waves
    for (int t = threadIdx.x; t < CH; t += 256) {
      const int g = ch * CH + t;
      if (g < Nq) {
        const float v = stage[par][t];
        if (v > MINSC) {
          const int pos = atomicAdd(&cnt, 1);
          if (pos < CAP) { cs[pos] = v; ci[pos] = g; }
        }
      }
    }
    __syncthreads();                            // stage reusable next round
  }

  int nc = cnt; if (nc > CAP) nc = CAP;
  for (int t = threadIdx.x; t < CAP; t += 256)
    if (t >= nc) { cs[t] = -1.0f; ci[t] = 0; }
  __syncthreads();

  // Bitonic sort, descending by score (index payload).
  for (int k = 2; k <= CAP; k <<= 1) {
    for (int j = k >> 1; j > 0; j >>= 1) {
      for (int t = threadIdx.x; t < CAP; t += 256) {
        const int x = t ^ j;
        if (x > t) {
          const bool up = (t & k) == 0;
          const float av = cs[t], bv = cs[x];
          if (up ? (av < bv) : (av > bv)) {
            cs[t] = bv; cs[x] = av;
            const int ti = ci[t]; ci[t] = ci[x]; ci[x] = ti;
          }
        }
      }
      __syncthreads();
    }
  }

  // Gather boxes for the top-KPRE candidates.
  for (int i = threadIdx.x; i < KPRE; i += 256) {
    const int idx = ci[i];
    const float* d = decoded + ((size_t)b * Nq + idx) * 4;
    const float x0 = d[0], y0 = d[1], x1 = d[2], y1 = d[3];
    bx0[i] = x0; by0[i] = y0; bx1[i] = x1; by1[i] = y1;
    barea[i] = (x1 - x0) * (y1 - y0);
    sup[i] = 0;
  }
  __syncthreads();

  // Greedy NMS (sorted desc => sequential over i, parallel over j).
  for (int i = 0; i < KPRE; ++i) {
    const bool keep = (cs[i] > MINSC) && (sup[i] == 0);   // uniform
    if (keep) {
      const float ax0 = bx0[i], ay0 = by0[i], ax1 = bx1[i], ay1 = by1[i];
      const float aa  = barea[i];
      for (int j = threadIdx.x; j < KPRE; j += 256) {
        if (j > i) {
          const float ix0 = fmaxf(ax0, bx0[j]), iy0 = fmaxf(ay0, by0[j]);
          const float ix1 = fminf(ax1, bx1[j]), iy1 = fminf(ay1, by1[j]);
          const float iw = fmaxf(ix1 - ix0, 0.0f), ih = fmaxf(iy1 - iy0, 0.0f);
          const float inter = iw * ih;
          const float iou = inter / (aa + barea[j] - inter + 1e-10f);
          if (iou > MAXIOU) sup[j] = 1;
        }
      }
    }
    __syncthreads();
  }

  for (int i = threadIdx.x; i < KPRE; i += 256) {
    const float v = cs[i];
    keptScore[(size_t)bc * KPRE + i] = (v > MINSC && sup[i] == 0) ? v : 0.0f;
    keptIdx  [(size_t)bc * KPRE + i] = ci[i];
  }
}

// =====================================================================
// Kernel 3: per-batch global top-200 over 80*300 kept scores + output.
// =====================================================================
__global__ __launch_bounds__(256) void final_topk_kernel(
    const float* __restrict__ keptScore, const int* __restrict__ keptIdx,
    const float* __restrict__ decoded,
    float* __restrict__ outBoxes, int* __restrict__ outLabels,
    float* __restrict__ outScores)
{
  __shared__ float cs[CAP2];
  __shared__ int   ci[CAP2];
  __shared__ int   cnt;

  const int b = blockIdx.x;
  if (threadIdx.x == 0) cnt = 0;
  __syncthreads();

  const float* ks = keptScore + (size_t)b * NCLS * KPRE;
  for (int t = threadIdx.x; t < NCLS * KPRE; t += 256) {
    const float v = ks[t];
    if (v > 0.0f) {
      const int pos = atomicAdd(&cnt, 1);
      if (pos < CAP2) { cs[pos] = v; ci[pos] = t; }
    }
  }
  __syncthreads();
  int nc = cnt; if (nc > CAP2) nc = CAP2;
  for (int t = threadIdx.x; t < CAP2; t += 256)
    if (t >= nc) { cs[t] = -1.0f; ci[t] = 0; }
  __syncthreads();

  for (int k = 2; k <= CAP2; k <<= 1) {
    for (int j = k >> 1; j > 0; j >>= 1) {
      for (int t = threadIdx.x; t < CAP2; t += 256) {
        const int x = t ^ j;
        if (x > t) {
          const bool up = (t & k) == 0;
          const float av = cs[t], bv = cs[x];
          if (up ? (av < bv) : (av > bv)) {
            cs[t] = bv; cs[x] = av;
            const int ti = ci[t]; ci[t] = ci[x]; ci[x] = ti;
          }
        }
      }
      __syncthreads();
    }
  }

  for (int t = threadIdx.x; t < TOPK; t += 256) {
    const float sc = cs[t];
    float o0 = 0.0f, o1 = 0.0f, o2 = 1.0f, o3 = 1.0f, osc = 0.0f;
    int lab = 0;
    if (sc > 0.0f) {
      const int f = ci[t];
      const int c = f / KPRE;
      const int n = keptIdx[(size_t)b * NCLS * KPRE + f];
      const float* d = decoded + ((size_t)b * Nq + n) * 4;
      o0 = d[0]; o1 = d[1]; o2 = d[2]; o3 = d[3];
      lab = c + 1; osc = sc;
    }
    float* ob = outBoxes + ((size_t)b * TOPK + t) * 4;
    ob[0] = o0; ob[1] = o1; ob[2] = o2; ob[3] = o3;
    outLabels[b * TOPK + t] = lab;
    outScores[b * TOPK + t] = osc;
  }
}

// =====================================================================
extern "C" void kernel_launch(void* const* d_in, const int* in_sizes, int n_in,
                              void* d_out, int out_size, void* d_ws, size_t ws_size,
                              hipStream_t stream)
{
  (void)in_sizes; (void)n_in; (void)out_size; (void)ws_size;

  const float* locs   = (const float*)d_in[0];  // predicted_boxes [B,N,4]
  const float* scores = (const float*)d_in[1];  // predicted_scores [B,N,81]
  const float* priors = (const float*)d_in[2];  // prior_boxes [N,4]

  // Workspace layout (~82 MB):
  char* ws = (char*)d_ws;
  float* decoded   = (float*)ws;                                   // B*N*4 f32
  size_t off = (size_t)TOT * 4 * sizeof(float);
  float* probsT    = (float*)(ws + off);                           // B*80*N f32
  off += (size_t)Bq * NCLS * Nq * sizeof(float);
  float* keptScore = (float*)(ws + off);                           // B*80*KPRE f32
  off += (size_t)Bq * NCLS * KPRE * sizeof(float);
  int*   keptIdx   = (int*)(ws + off);                             // B*80*KPRE i32

  // Output: boxes [B,200,4] f32 | labels [B,200] i32 | scores [B,200] f32
  float* outBoxes  = (float*)d_out;
  int*   outLabels = (int*)((float*)d_out + (size_t)Bq * TOPK * 4);
  float* outScores = (float*)d_out + (size_t)Bq * TOPK * 4 + (size_t)Bq * TOPK;

  softmax_decode_kernel<<<TOT / 128, 256, 0, stream>>>(scores, locs, priors,
                                                       probsT, decoded);
  select_nms_kernel<<<Bq * NCLS, 256, 0, stream>>>(probsT, decoded,
                                                   keptScore, keptIdx);
  final_topk_kernel<<<Bq, 256, 0, stream>>>(keptScore, keptIdx, decoded,
                                            outBoxes, outLabels, outScores);
}
